// GraphConvolution_1632087573322
// MI455X (gfx1250) — matrix operand order repaired and checked by reading, not compiled, not run on hardware
//
#include <hip/hip_runtime.h>
#include <hip/hip_bf16.h>
#include <math.h>

typedef __attribute__((ext_vector_type(16))) _Float16 v16h;
typedef __attribute__((ext_vector_type(8)))  float    v8f;

#define N_NODES 50000
#define DIM 64
#define M_TILES (N_NODES / 16)       // 3125, exact
#define MIN_NORM 1e-15f
#define ART_CLIP (1.0f - 1e-7f)
#define BALL_MAXNORM (1.0f - 4e-3f)  // (1-eps)/sqrt(c), c=1
#define B_TILES 8                    // 2 K-steps x 4 N-tiles
#define B_TILE_ELEMS 512             // 32 lanes x 16 halves

__device__ __forceinline__ float artanh_clip(float z) {
    z = fminf(fmaxf(z, -ART_CLIP), ART_CLIP);
    return atanhf(z);
}

// ---------------------------------------------------------------------------
// One-shot prep: repack W (f32 [out,in]) into the WMMA B-operand lane layout
// as f16 hi + f16 residual planes. Tile b = ksIdx*4 + t; within a tile,
// lane l owns column n = t*16 + (l&15) and K = ks + (e>=8?16:0) + (l>=16?8:0) + (e&7).
// ---------------------------------------------------------------------------
__global__ __launch_bounds__(256)
void wprep_kernel(const float* __restrict__ W,
                  _Float16* __restrict__ whi,
                  _Float16* __restrict__ wlo) {
    int i = blockIdx.x * 256 + threadIdx.x;          // 0 .. 4095
    if (i >= B_TILES * B_TILE_ELEMS) return;
    const int b = i >> 9;
    const int l = (i >> 4) & 31;
    const int e = i & 15;
    const int K = (b >> 2) * 32 + ((e >= 8) ? 16 : 0) + ((l >= 16) ? 8 : 0) + (e & 7);
    const int n = (b & 3) * 16 + (l & 15);
    const float f = W[n * DIM + K];
    const _Float16 h = (_Float16)f;
    whi[i] = h;
    wlo[i] = (_Float16)(f - (float)h);
}

// ---------------------------------------------------------------------------
// Kernel A: hidden_e = scale(x) * (x @ W^T), scale = artanh(|x|)/|x|
// (exact algebraic fusion of logmap0(mobius_matvec(W,x)) at c=1)
// ---------------------------------------------------------------------------
__global__ __launch_bounds__(128)
void hgcn_linear_logmap_kernel(const float* __restrict__ x,
                               const _Float16* __restrict__ whi,
                               const _Float16* __restrict__ wlo,
                               float* __restrict__ hidden_e) {
    const int lane  = threadIdx.x & 31;
    const int mtile = blockIdx.x * 4 + (threadIdx.x >> 5);
    if (mtile >= M_TILES) return;
    const int row0 = mtile * 16;

    // ---- per-row |x| and the fused artanh scale --------------------------
    const float* xr = x + (size_t)(row0 + (lane & 15)) * DIM + ((lane < 16) ? 0 : 32);
    float ss = 0.f;
#pragma unroll
    for (int j = 0; j < 32; j += 4) {
        float4 v = *(const float4*)(xr + j);
        ss += v.x * v.x + v.y * v.y + v.z * v.z + v.w * v.w;
    }
    ss += __shfl_xor(ss, 16);
    const float xn    = fmaxf(sqrtf(ss), MIN_NORM);
    const float scale = artanh_clip(xn) / xn;   // lane m (m<16) holds row m's scale

    // ---- WMMA GEMM -------------------------------------------------------
    v8f acc[4] = {};
    const int mA     = lane & 15;               // A row owned by this lane
    const int kHalfA = (lane >> 4) * 8;         // ISA 16-bit A layout K sub-base

#pragma unroll
    for (int ksIdx = 0; ksIdx < 2; ++ksIdx) {
        const int ks = ksIdx * 32;
        v16h ahi, alo;
#pragma unroll
        for (int v = 0; v < 8; ++v) {
            const int kA = ((v >= 4) ? 16 : 0) + kHalfA + (v & 3) * 2;
            float2 f = *(const float2*)(x + (size_t)(row0 + mA) * DIM + ks + kA);
            _Float16 h0 = (_Float16)f.x, h1 = (_Float16)f.y;
            ahi[2 * v]     = h0;               ahi[2 * v + 1]     = h1;
            alo[2 * v]     = (_Float16)(f.x - (float)h0);
            alo[2 * v + 1] = (_Float16)(f.y - (float)h1);
        }
#pragma unroll
        for (int t = 0; t < 4; ++t) {
            const size_t boff = (size_t)(ksIdx * 4 + t) * B_TILE_ELEMS + lane * 16;
            const v16h bhi = *(const v16h*)(whi + boff);   // 32B contiguous per lane
            const v16h blo = *(const v16h*)(wlo + boff);
            acc[t] = __builtin_amdgcn_wmma_f32_16x16x32_f16(
                false, ahi, false, bhi, (short)0, acc[t], false, false);
            acc[t] = __builtin_amdgcn_wmma_f32_16x16x32_f16(
                false, alo, false, bhi, (short)0, acc[t], false, false);
            acc[t] = __builtin_amdgcn_wmma_f32_16x16x32_f16(
                false, ahi, false, blo, (short)0, acc[t], false, false);
        }
    }

    // ---- scale rows and store --------------------------------------------
#pragma unroll
    for (int t = 0; t < 4; ++t) {
#pragma unroll
        for (int r = 0; r < 8; ++r) {
            const int m  = r + ((lane >= 16) ? 8 : 0);
            const float sc = __shfl(scale, m);
            hidden_e[(size_t)(row0 + m) * DIM + t * 16 + (lane & 15)] = sc * acc[t][r];
        }
    }
}

// ---------------------------------------------------------------------------
__global__ __launch_bounds__(256)
void zero_kernel(float* __restrict__ p, int n) {
    int i = blockIdx.x * 256 + threadIdx.x;
    if (i < n) p[i] = 0.0f;
}

// ---------------------------------------------------------------------------
// Edge scatter: support[row[e]] += ew[e] * hidden_e[col[e]]
// ---------------------------------------------------------------------------
__global__ __launch_bounds__(256)
void scatter_kernel(const float* __restrict__ hidden_e,
                    const int* __restrict__ rows,
                    const int* __restrict__ cols,
                    const float* __restrict__ ew,
                    float* __restrict__ support, int nE) {
    int i = blockIdx.x * 256 + threadIdx.x;
    int e = i >> 6;
    int f = i & 63;
    if (e >= nE) return;
    const int   c = cols[e];
    const int   r = rows[e];
    const float v = ew[e] * hidden_e[(size_t)c * DIM + f];
    atomicAdd(&support[(size_t)r * DIM + f], v);
}

// ---------------------------------------------------------------------------
// Finalize in-place: expmap0 -> logmap0(clip) -> relu -> expmap0 -> proj
// ---------------------------------------------------------------------------
__global__ __launch_bounds__(128)
void finalize_kernel(float* __restrict__ out) {
    const int node = blockIdx.x * 4 + (threadIdx.x >> 5);
    const int lane = threadIdx.x & 31;
    if (node >= N_NODES) return;
    float* rowp = out + (size_t)node * DIM + lane * 2;
    float2 s = *(const float2*)rowp;

    float ss = s.x * s.x + s.y * s.y;
#pragma unroll
    for (int m = 16; m >= 1; m >>= 1) ss += __shfl_xor(ss, m);
    const float n1 = fmaxf(sqrtf(ss), MIN_NORM);
    const float t1 = tanhf(n1) / n1;
    float2 p = {s.x * t1, s.y * t1};

    const float pn = fmaxf(tanhf(n1), MIN_NORM);       // |p|
    const float lc = artanh_clip(pn) / pn;
    float2 xt = {fmaxf(p.x * lc, 0.0f), fmaxf(p.y * lc, 0.0f)};

    float ts = xt.x * xt.x + xt.y * xt.y;
#pragma unroll
    for (int m = 16; m >= 1; m >>= 1) ts += __shfl_xor(ts, m);
    const float n2 = fmaxf(sqrtf(ts), MIN_NORM);
    const float t2 = tanhf(n2) / n2;
    float2 o = {xt.x * t2, xt.y * t2};

    const float on = fmaxf(tanhf(n2), MIN_NORM);       // |o|
    if (on > BALL_MAXNORM) {
        const float k = BALL_MAXNORM / on;
        o.x *= k; o.y *= k;
    }
    *(float2*)rowp = o;
}

// ---------------------------------------------------------------------------
extern "C" void kernel_launch(void* const* d_in, const int* in_sizes, int n_in,
                              void* d_out, int out_size, void* d_ws, size_t ws_size,
                              hipStream_t stream) {
    const float* x  = (const float*)d_in[0];   // [N, 64]
    const float* W  = (const float*)d_in[1];   // [64, 64]
    const float* ew = (const float*)d_in[2];   // [E]
    const int*   ei = (const int*)d_in[3];     // [2, E] int32
    const int nE = in_sizes[2];
    const int* rows = ei;
    const int* cols = ei + nE;

    float*    wsf      = (float*)d_ws;
    float*    hidden_e = wsf;                                  // N*64 f32
    _Float16* whi      = (_Float16*)(wsf + (size_t)N_NODES * DIM);
    _Float16* wlo      = whi + B_TILES * B_TILE_ELEMS;
    float*    out      = (float*)d_out;        // doubles as support accumulator

    // 0: repack W into WMMA B lane layout (f16 hi/lo planes)
    wprep_kernel<<<(B_TILES * B_TILE_ELEMS + 255) / 256, 256, 0, stream>>>(W, whi, wlo);

    // A: WMMA GEMM + fused logmap0(mobius_matvec) scale
    hgcn_linear_logmap_kernel<<<(M_TILES + 3) / 4, 128, 0, stream>>>(x, whi, wlo, hidden_e);

    // zero support (d_out) then scatter-accumulate edges
    const int ntot = N_NODES * DIM;
    zero_kernel<<<(ntot + 255) / 256, 256, 0, stream>>>(out, ntot);
    const int work = nE * DIM;                 // 51.2M element-ops
    scatter_kernel<<<(work + 255) / 256, 256, 0, stream>>>(hidden_e, rows, cols, ew, out, nE);

    // expmap0 / relu(logmap0) / expmap0 / proj, in place
    finalize_kernel<<<(N_NODES + 3) / 4, 128, 0, stream>>>(out);
}
